// DigitConvolutionalModel_62826781606401
// MI455X (gfx1250) — matrix operand order repaired and checked by reading, compiled, not run on hardware
//
#include <hip/hip_runtime.h>

typedef _Float16 half_t;
typedef __attribute__((ext_vector_type(16))) _Float16 v16h;
typedef __attribute__((ext_vector_type(8)))  float    v8f;

#define BROWS   64                    // batch rows per block (4 m-tiles)
#define KP      704                   // GEMM1 K padded 676 -> 704 (mult of 32)
#define NK      (KP / 32)             // 22 k-steps
#define FSTR    712                   // featH LDS row stride (halfs): 16B aligned, bank-spread
#define HSTR    264                   // hH LDS row stride (halfs)
#define BSTR    40                    // staged-B row stride (halfs): 16B aligned, conflict-free
#define BBUF_H  (256 * BSTR)          // halfs per B buffer
#define FEAT_BYTES (BROWS * FSTR * 2) // 91136
#define SMEM_BYTES (FEAT_BYTES + 2 * BBUF_H * 2)  // 91136 + 40960 = 132096
#define W1H_ELEMS (256 * KP)
#define W2H_ELEMS (16 * 256)

union U16 { uint4 q[2]; v16h h; };

// ---- weight prep: f32 -> f16, padded/transposed for WMMA B-fragment loads ----
__global__ __launch_bounds__(256) void prep_weights(const float* __restrict__ w1,
                                                    const float* __restrict__ w2,
                                                    half_t* __restrict__ w1h,
                                                    half_t* __restrict__ w2h) {
  int idx = blockIdx.x * 256 + threadIdx.x;
  if (idx < W1H_ELEMS) {
    int n = idx / KP, k = idx - n * KP;
    float v = (k < 676) ? w1[k * 256 + n] : 0.0f;   // w1 is [K=676][N=256]
    w1h[n * KP + k] = (half_t)v;
  } else {
    int j = idx - W1H_ELEMS;
    if (j < W2H_ELEMS) {
      int n = j / 256, k = j - n * 256;
      float v = (n < 10) ? w2[k * 10 + n] : 0.0f;   // w2 is [K=256][N=10]
      w2h[n * 256 + k] = (half_t)v;
    }
  }
}

// ---- fused conv -> GEMM1(relu) -> GEMM2 ----
__global__ __launch_bounds__(256) void fused_digit_mlp(
    const float* __restrict__ x, const float* __restrict__ wconv,
    const half_t* __restrict__ w1h, const float* __restrict__ b1,
    const half_t* __restrict__ w2h, const float* __restrict__ b2,
    float* __restrict__ out) {
  __shared__ __align__(16) unsigned char smem[SMEM_BYTES];
  half_t* featH = (half_t*)smem;                 // [64][FSTR] f16 conv features
  half_t* bS    = (half_t*)(smem + FEAT_BYTES);  // 2 x [256][BSTR] staged w1 chunks
  half_t* hH    = bS;                            // [64][HSTR] f16 h (aliases bS after K loop)

  const int tid  = threadIdx.x;
  const int lane = tid & 31;
  const int wv   = tid >> 5;        // wave 0..7
  const int sel  = lane >> 4;       // half-wave select
  const int ln   = lane & 15;
  const long r0  = (long)blockIdx.x * BROWS;

  float wc[9];
#pragma unroll
  for (int i = 0; i < 9; ++i) wc[i] = wconv[i];

  // ---- 1) conv 3x3 (valid): one (image, conv-row) task per thread iteration.
  // 21 float4 loads for the 3 input rows, 26 outputs from registers, 13 b32 LDS stores.
  const int NTASK = BROWS * 26;
#pragma unroll 1
  for (int t = tid; t < NTASK; t += 256) {
    const int m = t / 26, rr = t - m * 26;
    const float* xrow = x + (r0 + m) * 784 + rr * 28;
    float a[3][28];
#pragma unroll
    for (int dr = 0; dr < 3; ++dr)
#pragma unroll
      for (int q = 0; q < 7; ++q)
        *(float4*)&a[dr][4 * q] = *(const float4*)(xrow + dr * 28 + 4 * q);

    half_t* drow = featH + m * FSTR + rr * 26;
#pragma unroll
    for (int p = 0; p < 13; ++p) {
      union { half_t h[2]; unsigned u; } pk;
#pragma unroll
      for (int e = 0; e < 2; ++e) {
        const int j = 2 * p + e;
        float v = a[0][j] * wc[0] + a[0][j + 1] * wc[1] + a[0][j + 2] * wc[2]
                + a[1][j] * wc[3] + a[1][j + 1] * wc[4] + a[1][j + 2] * wc[5]
                + a[2][j] * wc[6] + a[2][j + 1] * wc[7] + a[2][j + 2] * wc[8];
        pk.h[e] = (half_t)v;
      }
      *(unsigned*)(drow + 2 * p) = pk.u;
    }
  }
  // zero-fill K padding columns 676..703
  for (int i = tid; i < BROWS * (KP - 676); i += 256) {
    int m = i / (KP - 676), k = 676 + (i - m * (KP - 676));
    featH[m * FSTR + k] = (half_t)0.0f;
  }

  // ---- prologue: stage w1 k-chunk 0 into LDS buffer 0 (thread t <-> column t)
  const half_t* wrow = w1h + (long)tid * KP;
  {
    uint4 t0 = *(const uint4*)(wrow + 0);
    uint4 t1 = *(const uint4*)(wrow + 8);
    uint4 t2 = *(const uint4*)(wrow + 16);
    uint4 t3 = *(const uint4*)(wrow + 24);
    half_t* d = bS + tid * BSTR;
    *(uint4*)(d + 0) = t0;  *(uint4*)(d + 8)  = t1;
    *(uint4*)(d + 16) = t2; *(uint4*)(d + 24) = t3;
  }
  __syncthreads();

  // ---- 2) GEMM1: 64x704 @ 704x256, per-wave 2Mx4N tile, double-buffered B in LDS
  const int mg = wv & 1;            // m-pair select: rows mg*32 .. mg*32+31
  const int ng = wv >> 1;           // n-group: cols ng*64 .. ng*64+63
  const int koffA = sel * 8;        // A layout: upper half-wave shifts K by 8
  v8f acc[2][4] = {};

  for (int ks = 0; ks < NK; ++ks) {
    const int kk = ks * 32;
    const bool more = (ks + 1 < NK);
    uint4 t0 = {}, t1 = {}, t2 = {}, t3 = {};
    if (more) {                      // issue next chunk's global loads early
      const half_t* g = wrow + kk + 32;
      t0 = *(const uint4*)(g + 0);  t1 = *(const uint4*)(g + 8);
      t2 = *(const uint4*)(g + 16); t3 = *(const uint4*)(g + 24);
    }
    const half_t* bb = bS + (ks & 1) * BBUF_H;

    U16 A[2];
#pragma unroll
    for (int mi = 0; mi < 2; ++mi) {
      const half_t* ap = featH + (mg * 32 + mi * 16 + ln) * FSTR + kk + koffA;
      A[mi].q[0] = *(const uint4*)(ap);
      A[mi].q[1] = *(const uint4*)(ap + 16);
    }
    U16 Bf[4];
#pragma unroll
    for (int ni = 0; ni < 4; ++ni) {
      const half_t* bp = bb + (ng * 64 + ni * 16 + ln) * BSTR + sel * 8;
      Bf[ni].q[0] = *(const uint4*)(bp);
      Bf[ni].q[1] = *(const uint4*)(bp + 16);
    }
#pragma unroll
    for (int ni = 0; ni < 4; ++ni)
#pragma unroll
      for (int mi = 0; mi < 2; ++mi)
        acc[mi][ni] = __builtin_amdgcn_wmma_f32_16x16x32_f16(
            false, A[mi].h, false, Bf[ni].h, (short)0, acc[mi][ni], false, false);

    if (more) {                      // drain prefetch into the other buffer
      half_t* d = bS + ((ks + 1) & 1) * BBUF_H + tid * BSTR;
      *(uint4*)(d + 0) = t0;  *(uint4*)(d + 8)  = t1;
      *(uint4*)(d + 16) = t2; *(uint4*)(d + 24) = t3;
    }
    __syncthreads();
  }

  // ---- epilogue: +b1, relu, f16 h tile into LDS (C layout: VGPR r -> M = r + 8*sel)
#pragma unroll
  for (int ni = 0; ni < 4; ++ni) {
    const int n = ng * 64 + ni * 16 + ln;
    const float bias = b1[n];
#pragma unroll
    for (int mi = 0; mi < 2; ++mi) {
#pragma unroll
      for (int r = 0; r < 8; ++r) {
        float v = acc[mi][ni][r] + bias;
        v = v > 0.0f ? v : 0.0f;
        hH[(mg * 32 + mi * 16 + r + sel * 8) * HSTR + n] = (half_t)v;
      }
    }
  }
  __syncthreads();

  // ---- 3) GEMM2: h(64x256) @ w2(256x10), waves 0..3, one 16-row tile each
  if (wv < 4) {
    const int grow = wv * 16 + ln;
    v8f a2 = {};
    for (int kk = 0; kk < 256; kk += 32) {
      U16 ua, ub;
      const half_t* ap = hH + grow * HSTR + kk + koffA;
      ua.q[0] = *(const uint4*)(ap);
      ua.q[1] = *(const uint4*)(ap + 16);
      const half_t* bp = w2h + ln * 256 + kk + sel * 8;
      ub.q[0] = *(const uint4*)(bp);
      ub.q[1] = *(const uint4*)(bp + 16);
      a2 = __builtin_amdgcn_wmma_f32_16x16x32_f16(
          false, ua.h, false, ub.h, (short)0, a2, false, false);
    }
    if (ln < 10) {
      const float bias = b2[ln];
#pragma unroll
      for (int r = 0; r < 8; ++r) {
        const long row = r0 + wv * 16 + r + 8 * sel;
        out[row * 10 + ln] = a2[r] + bias;
      }
    }
  }
}

extern "C" void kernel_launch(void* const* d_in, const int* in_sizes, int n_in,
                              void* d_out, int out_size, void* d_ws, size_t ws_size,
                              hipStream_t stream) {
  (void)in_sizes; (void)n_in; (void)out_size; (void)ws_size;
  const float* x     = (const float*)d_in[0];  // (65536, 784)
  const float* wconv = (const float*)d_in[1];  // (3,3)
  const float* w1    = (const float*)d_in[2];  // (676, 256)
  const float* b1    = (const float*)d_in[3];  // (256,)
  const float* w2    = (const float*)d_in[4];  // (256, 10)
  const float* b2    = (const float*)d_in[5];  // (10,)
  float* out = (float*)d_out;

  half_t* w1h = (half_t*)d_ws;
  half_t* w2h = w1h + W1H_ELEMS;

  const int prep_items = W1H_ELEMS + W2H_ELEMS;
  prep_weights<<<(prep_items + 255) / 256, 256, 0, stream>>>(w1, w2, w1h, w2h);
  fused_digit_mlp<<<65536 / BROWS, 256, 0, stream>>>(x, wconv, w1h, b1, w2h, b2, out);
}
